// SlotPool_87565793231224
// MI455X (gfx1250) — compile-verified, matrix operand order checked by reading, not run on hardware
//
#include <hip/hip_runtime.h>
#include <hip/hip_bf16.h>
#include <math.h>

// ---------------------------------------------------------------------------
// SlotPool on gfx1250 (MI455X): f16 WMMA for all GEMMs, TDM (tensor_load_to_lds)
// for plain 2D tile staging where available.
// ---------------------------------------------------------------------------

#define BATCH 32
#define NTOK  4096
#define CH    768
#define NSLOT 8
#define HID   1536
#define ROWS  (BATCH * NSLOT)      // 256 slot rows
#define XROWS (BATCH * NTOK)       // 131072 token rows
#define SCALE 0.03608439182435161f // 768^-0.5
#define EPS_LN 1e-5f
#define EPS_ATTN 1e-8f

typedef _Float16 v8h  __attribute__((ext_vector_type(8)));
typedef _Float16 v16h __attribute__((ext_vector_type(16)));
typedef float    v8f  __attribute__((ext_vector_type(8)));

#if defined(__gfx1250__) && __has_builtin(__builtin_amdgcn_tensor_load_to_lds) && \
    __has_builtin(__builtin_amdgcn_s_wait_tensorcnt)
#define USE_TDM 1
#else
#define USE_TDM 0
#endif

#if USE_TDM
typedef unsigned int u32x4 __attribute__((ext_vector_type(4)));
typedef int          i32x4 __attribute__((ext_vector_type(4)));
typedef int          i32x8 __attribute__((ext_vector_type(8)));

// TDM load of a 128-row x 64-byte tile (f16 128x32) into LDS with a 16-byte
// pad after every row (LDS pitch 80B = 40 halves). data_size = 4B units.
// D# packing per CDNA5 ISA.
__device__ __forceinline__ void tdm_load_tile(unsigned lds_off,
                                              unsigned long long gaddr,
                                              unsigned stride_u4) {
  u32x4 g0;
  g0[0] = 1u;                                   // count=1 (valid user D#)
  g0[1] = lds_off;                              // lds_addr (bytes)
  g0[2] = (unsigned)gaddr;                      // global_addr[31:0]
  g0[3] = (unsigned)((gaddr >> 32) & 0x01FFFFFFu) | (2u << 30);  // [56:32]|type=2
  i32x8 g1;
  g1[0] = (int)((2u << 16)      // data_size = 4B
                | (1u << 20)    // pad_enable
                | (3u << 22)    // pad_interval: 16 DWORDs (64B)
                | (3u << 25));  // pad_amount: 4 DWORDs (16B)
  g1[1] = (int)(16u << 16);     // tensor_dim0 = 16 units (bits 63:48)
  g1[2] = (int)(128u << 16);    // tensor_dim1 = 128 (bits 95:80)
  g1[3] = (int)(16u << 16);     // tile_dim0 = 16 units (bits 127:112)
  g1[4] = (int)128u;            // tile_dim1 = 128, tile_dim2 = 0
  g1[5] = (int)stride_u4;       // tensor_dim0_stride low32 (4B units)
  g1[6] = 0;
  g1[7] = 0;
  i32x4 z4;
  z4[0] = z4[1] = z4[2] = z4[3] = 0;
#if __clang_major__ >= 23
  i32x8 z8;
#pragma unroll
  for (int i = 0; i < 8; ++i) z8[i] = 0;
  __builtin_amdgcn_tensor_load_to_lds(g0, g1, z4, z4, z8, 0);
#else
  __builtin_amdgcn_tensor_load_to_lds(g0, g1, z4, z4, 0);
#endif
}
#endif  // USE_TDM

__device__ __forceinline__ v8f wmma_f16(v16h a, v16h b, v8f c) {
  return __builtin_amdgcn_wmma_f32_16x16x32_f16(false, a, false, b, (short)0, c,
                                                false, false);
}

// Build a v16h fragment: chunk [k0, k0+8) and [16+k0, 16+k0+8) from a row base.
// Matches the documented 16-bit A layout (lanes 0-15: K=0..7 & 16..23,
// lanes 16-31: K=8..15 & 24..31); B assumed symmetric with N in place of M.
__device__ __forceinline__ v16h frag16(const _Float16* p, int k0) {
  v8h lo = *(const v8h*)(p + k0);
  v8h hi = *(const v8h*)(p + 16 + k0);
  return __builtin_shufflevector(lo, hi, 0, 1, 2, 3, 4, 5, 6, 7, 8, 9, 10, 11,
                                 12, 13, 14, 15);
}

__device__ __forceinline__ v8f zero8() {
  v8f z;
#pragma unroll
  for (int i = 0; i < 8; ++i) z[i] = 0.0f;
  return z;
}

// ---------------------------- utility kernels ------------------------------

__global__ void k_cast_f32_f16(const float* __restrict__ s,
                               _Float16* __restrict__ d, int n) {
  int i = blockIdx.x * blockDim.x + threadIdx.x;
  if (i < n) d[i] = (_Float16)s[i];
}

__global__ void k_zero_f16(_Float16* p, int n) {
  int i = blockIdx.x * blockDim.x + threadIdx.x;
  if (i < n) p[i] = (_Float16)0.0f;
}

__global__ void k_zero_f32(float* p, int n) {
  int i = blockIdx.x * blockDim.x + threadIdx.x;
  if (i < n) p[i] = 0.0f;
}

__global__ void k_init_slots(const float* __restrict__ mu,
                             float* __restrict__ slots) {
  int i = blockIdx.x * blockDim.x + threadIdx.x;
  if (i < ROWS * CH) slots[i] = mu[i % CH];
}

// per-row mean / rstd of x (for fused LayerNorm in proj_kv). One wave per row.
__global__ __launch_bounds__(256) void k_row_stats(const float* __restrict__ x,
                                                   float* __restrict__ mr) {
  int row  = blockIdx.x * 8 + (threadIdx.x >> 5);
  int lane = threadIdx.x & 31;
  const float* p = x + (size_t)row * CH;
  float s = 0.0f, q = 0.0f;
#pragma unroll
  for (int i = 0; i < CH / 32; ++i) {
    float v = p[lane + i * 32];
    s += v;
    q += v * v;
  }
#pragma unroll
  for (int m = 16; m >= 1; m >>= 1) {
    s += __shfl_xor(s, m);
    q += __shfl_xor(q, m);
  }
  if (lane == 0) {
    float mean = s * (1.0f / CH);
    float var  = q * (1.0f / CH) - mean * mean;
    mr[2 * row]     = mean;
    mr[2 * row + 1] = rsqrtf(var + EPS_LN);
  }
}

// LayerNorm of the 256 slot rows -> f16 normalized + raw f16 copy.
__global__ __launch_bounds__(256) void k_slot_ln(const float* __restrict__ in,
                                                 const float* __restrict__ w,
                                                 const float* __restrict__ b,
                                                 _Float16* __restrict__ outn,
                                                 _Float16* __restrict__ outraw) {
  int row  = blockIdx.x * 8 + (threadIdx.x >> 5);
  int lane = threadIdx.x & 31;
  const float* p = in + (size_t)row * CH;
  float vals[CH / 32];
  float s = 0.0f, q = 0.0f;
#pragma unroll
  for (int i = 0; i < CH / 32; ++i) {
    float v = p[lane + i * 32];
    vals[i] = v;
    s += v;
    q += v * v;
  }
#pragma unroll
  for (int m = 16; m >= 1; m >>= 1) {
    s += __shfl_xor(s, m);
    q += __shfl_xor(q, m);
  }
  float mean = s * (1.0f / CH);
  float rstd = rsqrtf(q * (1.0f / CH) - mean * mean + EPS_LN);
#pragma unroll
  for (int i = 0; i < CH / 32; ++i) {
    int c = lane + i * 32;
    outn[(size_t)row * CH + c] =
        (_Float16)((vals[i] - mean) * rstd * w[c] + b[c]);
    outraw[(size_t)row * CH + c] = (_Float16)vals[i];
  }
}

// --------------- fused LN + K/V projection (the big GEMMs) -----------------
// k = LN(x)@kw^T + kb ; v = LN(x)@vw^T + vb, both f16 out.
__global__ __launch_bounds__(256) void k_proj_kv(
    const float* __restrict__ x, const float* __restrict__ mr,
    const float* __restrict__ lnw, const float* __restrict__ lnb,
    const _Float16* __restrict__ kw, const _Float16* __restrict__ vw,
    const float* __restrict__ kb, const float* __restrict__ vb,
    _Float16* __restrict__ kout, _Float16* __restrict__ vout) {
  __shared__ _Float16 lA[128 * 40];
  __shared__ _Float16 lK[128 * 40];
  __shared__ _Float16 lV[128 * 40];
  __shared__ float s_w[CH];
  __shared__ float s_b[CH];
  const int tid = threadIdx.x;
  const int m0  = blockIdx.x * 128;
  const int n0  = blockIdx.y * 128;
  for (int i = tid; i < CH; i += 256) {
    s_w[i] = lnw[i];
    s_b[i] = lnb[i];
  }
  const int wid = tid >> 5, lane = tid & 31, lm = lane & 15,
            k0 = (lane >> 4) * 8;
  v8f acck[8], accv[8];
#pragma unroll
  for (int t = 0; t < 8; ++t) {
    acck[t] = zero8();
    accv[t] = zero8();
  }
  __syncthreads();
  for (int kk = 0; kk < CH; kk += 32) {
    // A tile: x -> LN -> f16 (128 rows x 32 K); data transform -> VGPR path
    for (int idx = tid; idx < 512; idx += 256) {
      int r = idx >> 2, ko = (idx & 3) * 8;
      int gr = m0 + r;
      const float4* px = (const float4*)(x + (size_t)gr * CH + kk + ko);
      float4 f0 = px[0], f1 = px[1];
      if (kk + 32 < CH) __builtin_prefetch(px + 8, 0, 1);
      float mean = mr[2 * gr], rstd = mr[2 * gr + 1];
      float wv[8] = {f0.x, f0.y, f0.z, f0.w, f1.x, f1.y, f1.z, f1.w};
      _Float16* dst = &lA[r * 40 + ko];
#pragma unroll
      for (int i = 0; i < 8; ++i) {
        int c = kk + ko + i;
        dst[i] = (_Float16)((wv[i] - mean) * rstd * s_w[c] + s_b[c]);
      }
    }
    // W tiles: plain 2D copies -> Tensor Data Mover when available
#if USE_TDM
    if (wid == 0) {
      tdm_load_tile((unsigned)(size_t)&lK[0],
                    (unsigned long long)(size_t)(kw + (size_t)n0 * CH + kk),
                    CH / 2);
      __builtin_amdgcn_s_wait_tensorcnt(0);
    } else if (wid == 1) {
      tdm_load_tile((unsigned)(size_t)&lV[0],
                    (unsigned long long)(size_t)(vw + (size_t)n0 * CH + kk),
                    CH / 2);
      __builtin_amdgcn_s_wait_tensorcnt(0);
    }
#else
    for (int idx = tid; idx < 512; idx += 256) {
      int r = idx >> 2, ko = (idx & 3) * 8;
      int gn = n0 + r;
      *(v8h*)&lK[r * 40 + ko] = *(const v8h*)(kw + (size_t)gn * CH + kk + ko);
      *(v8h*)&lV[r * 40 + ko] = *(const v8h*)(vw + (size_t)gn * CH + kk + ko);
    }
#endif
    __syncthreads();
    v16h a = frag16(&lA[(wid * 16 + lm) * 40], k0);
#pragma unroll
    for (int nt = 0; nt < 8; ++nt) {
      v16h bk = frag16(&lK[(nt * 16 + lm) * 40], k0);
      v16h bv = frag16(&lV[(nt * 16 + lm) * 40], k0);
      acck[nt] = wmma_f16(a, bk, acck[nt]);
      accv[nt] = wmma_f16(a, bv, accv[nt]);
    }
    __syncthreads();
  }
  const int hi = lane >> 4;
#pragma unroll
  for (int nt = 0; nt < 8; ++nt) {
    int col = n0 + nt * 16 + lm;
    float kbv = kb[col], vbv = vb[col];
#pragma unroll
    for (int j = 0; j < 8; ++j) {
      int row  = m0 + wid * 16 + (hi ? j + 8 : j);
      size_t o = (size_t)row * CH + col;
      kout[o]  = (_Float16)(acck[nt][j] + kbv);
      vout[o]  = (_Float16)(accv[nt][j] + vbv);
    }
  }
}

// ------------------- generic C = A @ W^T + bias WMMA GEMM ------------------
// mode 0: f16 out   1: GELU->f16   2: f32 out   3: f32 residual add
// mode 4: f16 scatter into zero-padded q buffer [B][16][CH]
__global__ __launch_bounds__(256) void k_gemm_xwT(
    const _Float16* __restrict__ A, const _Float16* __restrict__ W,
    const float* __restrict__ bias, int N, int K, int mode,
    float* __restrict__ out32, _Float16* __restrict__ out16) {
  __shared__ _Float16 lA[128 * 40];
  __shared__ _Float16 lW[128 * 40];
  const int tid = threadIdx.x;
  const int m0  = blockIdx.x * 128;
  const int n0  = blockIdx.y * 128;
  const int wid = tid >> 5, lane = tid & 31, lm = lane & 15,
            k0 = (lane >> 4) * 8;
  v8f acc[8];
#pragma unroll
  for (int t = 0; t < 8; ++t) acc[t] = zero8();
  for (int kk = 0; kk < K; kk += 32) {
#if USE_TDM
    if (wid == 0) {
      tdm_load_tile((unsigned)(size_t)&lA[0],
                    (unsigned long long)(size_t)(A + (size_t)m0 * K + kk),
                    K / 2);
      __builtin_amdgcn_s_wait_tensorcnt(0);
    } else if (wid == 1) {
      tdm_load_tile((unsigned)(size_t)&lW[0],
                    (unsigned long long)(size_t)(W + (size_t)n0 * K + kk),
                    K / 2);
      __builtin_amdgcn_s_wait_tensorcnt(0);
    }
#else
    for (int idx = tid; idx < 512; idx += 256) {
      int r = idx >> 2, ko = (idx & 3) * 8;
      *(v8h*)&lA[r * 40 + ko] =
          *(const v8h*)(A + (size_t)(m0 + r) * K + kk + ko);
      *(v8h*)&lW[r * 40 + ko] =
          *(const v8h*)(W + (size_t)(n0 + r) * K + kk + ko);
    }
#endif
    __syncthreads();
    v16h a = frag16(&lA[(wid * 16 + lm) * 40], k0);
#pragma unroll
    for (int nt = 0; nt < 8; ++nt) {
      v16h b  = frag16(&lW[(nt * 16 + lm) * 40], k0);
      acc[nt] = wmma_f16(a, b, acc[nt]);
    }
    __syncthreads();
  }
  const int hi = lane >> 4;
#pragma unroll
  for (int nt = 0; nt < 8; ++nt) {
    int col  = n0 + nt * 16 + lm;
    float bv = bias[col];
#pragma unroll
    for (int j = 0; j < 8; ++j) {
      int row = m0 + wid * 16 + (hi ? j + 8 : j);
      float v = acc[nt][j] + bv;
      if (mode == 0) {
        out16[(size_t)row * N + col] = (_Float16)v;
      } else if (mode == 1) {
        float g = 0.5f * v * (1.0f + erff(v * 0.70710678118654752f));
        out16[(size_t)row * N + col] = (_Float16)g;
      } else if (mode == 2) {
        out32[(size_t)row * N + col] = v;
      } else if (mode == 3) {
        size_t o = (size_t)row * N + col;
        out32[o] = out32[o] + v;
      } else {  // mode 4: q scatter, rows padded to 16 per batch
        size_t o = (size_t)((row >> 3) * 16 + (row & 7)) * CH + col;
        out16[o] = (_Float16)v;
      }
    }
  }
}

// ----------------- attention logits + softmax-over-slots -------------------
// attn[b,s,n] = softmax_s(q.k * SCALE) + eps ; rowsum[b,s] += sum_n attn
__global__ __launch_bounds__(256) void k_attn_logits(
    const _Float16* __restrict__ qpad, const _Float16* __restrict__ kmat,
    float* __restrict__ attn, float* __restrict__ rowsum) {
  const int b   = blockIdx.x;
  const int tid = threadIdx.x;
  const int wid = tid >> 5, lane = tid & 31, lm = lane & 15,
            k0 = (lane >> 4) * 8;
  const int n = blockIdx.y * 128 + wid * 16 + lm;
  const _Float16* qb = qpad + ((size_t)b * 16 + lm) * CH;
  const _Float16* kb = kmat + ((size_t)b * NTOK + n) * CH;
  v8f acc = zero8();
  for (int kk = 0; kk < CH; kk += 32) {
    v16h a = frag16(qb + kk, k0);
    v16h bfrag = frag16(kb + kk, k0);
    acc = wmma_f16(a, bfrag, acc);
  }
  // lanes 0-15 hold slot rows 0..7 for their column n -> lane-local softmax
  bool act = (lane < 16);
  float vals[8], e[8];
  float mx = -3.4e38f;
#pragma unroll
  for (int j = 0; j < 8; ++j) {
    vals[j] = acc[j] * SCALE;
    mx      = fmaxf(mx, vals[j]);
  }
  float den = 0.0f;
#pragma unroll
  for (int j = 0; j < 8; ++j) {
    e[j] = expf(vals[j] - mx);
    den += e[j];
  }
  float part[8];
#pragma unroll
  for (int j = 0; j < 8; ++j) {
    float aij = act ? (e[j] / den + EPS_ATTN) : 0.0f;
    part[j]   = aij;
    if (act) attn[((size_t)b * NSLOT + j) * NTOK + n] = aij;
  }
#pragma unroll
  for (int j = 0; j < 8; ++j) {
#pragma unroll
    for (int m = 16; m >= 1; m >>= 1) part[j] += __shfl_xor(part[j], m);
  }
  if (lane == 0) {
#pragma unroll
    for (int j = 0; j < 8; ++j) atomicAdd(&rowsum[b * NSLOT + j], part[j]);
  }
}

// --------------------- updates = (attn / rowsum) @ v -----------------------
__global__ __launch_bounds__(256) void k_attn_updates(
    const float* __restrict__ attn, const float* __restrict__ rowsum,
    const _Float16* __restrict__ vmat, float* __restrict__ upd32,
    _Float16* __restrict__ upd16) {
  __shared__ _Float16 lA[16 * 40];
  __shared__ _Float16 vT[128 * 40];
  __shared__ float sinv[NSLOT];
  const int b   = blockIdx.x;
  const int ct  = blockIdx.y;
  const int tid = threadIdx.x;
  const int wid = tid >> 5, lane = tid & 31, lm = lane & 15,
            k0 = (lane >> 4) * 8;
  if (tid < NSLOT) sinv[tid] = 1.0f / rowsum[b * NSLOT + tid];
  __syncthreads();
  v8f acc = zero8();
  for (int nb = 0; nb < NTOK; nb += 32) {
    if (tid < 64) {
      int s = tid >> 2, ko = (tid & 3) * 8;
      _Float16* dst = &lA[s * 40 + ko];
      if (s < NSLOT) {
        const float* pa = attn + ((size_t)b * NSLOT + s) * NTOK + nb + ko;
        float inv = sinv[s];
#pragma unroll
        for (int i = 0; i < 8; ++i) dst[i] = (_Float16)(pa[i] * inv);
      } else {
#pragma unroll
        for (int i = 0; i < 8; ++i) dst[i] = (_Float16)0.0f;
      }
    }
    {
      int n = tid >> 3, cc = (tid & 7) * 16;
      const _Float16* pv =
          vmat + ((size_t)b * NTOK + nb + n) * CH + ct * 128 + cc;
      v8h p0 = *(const v8h*)pv;
      v8h p1 = *(const v8h*)(pv + 8);
#pragma unroll
      for (int i = 0; i < 8; ++i) {
        vT[(cc + i) * 40 + n]     = p0[i];
        vT[(cc + 8 + i) * 40 + n] = p1[i];
      }
    }
    __syncthreads();
    v16h a = frag16(&lA[lm * 40], k0);
    v16h bfrag = frag16(&vT[(wid * 16 + lm) * 40], k0);
    acc = wmma_f16(a, bfrag, acc);
    __syncthreads();
  }
  if (lane < 16) {
    int c = ct * 128 + wid * 16 + lm;
#pragma unroll
    for (int j = 0; j < 8; ++j) {  // rows 0..7 are the valid slots
      size_t o = ((size_t)b * NSLOT + j) * CH + c;
      upd32[o] = acc[j];
      upd16[o] = (_Float16)acc[j];
    }
  }
}

// ------------------------------ GRU gates ----------------------------------
__global__ void k_gru(const float* __restrict__ gi, const float* __restrict__ gh,
                      float* __restrict__ slots) {
  int i = blockIdx.x * blockDim.x + threadIdx.x;
  if (i >= ROWS * CH) return;
  int row = i / CH, c = i % CH;
  size_t base = (size_t)row * (3 * CH) + c;
  float ir = gi[base], iz = gi[base + CH], in_ = gi[base + 2 * CH];
  float hr = gh[base], hz = gh[base + CH], hn = gh[base + 2 * CH];
  float r = 1.0f / (1.0f + expf(-(ir + hr)));
  float z = 1.0f / (1.0f + expf(-(iz + hz)));
  float n = tanhf(in_ + r * hn);
  float h = slots[i];
  slots[i] = (1.0f - z) * n + z * h;
}

// ------------------------------ max pool -----------------------------------
__global__ void k_maxpool(const float* __restrict__ slots,
                          float* __restrict__ out) {
  int i = blockIdx.x * blockDim.x + threadIdx.x;
  if (i >= BATCH * CH) return;
  int b = i / CH, c = i % CH;
  float m = -3.4e38f;
#pragma unroll
  for (int s = 0; s < NSLOT; ++s)
    m = fmaxf(m, slots[((size_t)b * NSLOT + s) * CH + c]);
  out[i] = m;
}

// ---------------------------------------------------------------------------
extern "C" void kernel_launch(void* const* d_in, const int* in_sizes, int n_in,
                              void* d_out, int out_size, void* d_ws,
                              size_t ws_size, hipStream_t stream) {
  (void)in_sizes; (void)n_in; (void)out_size; (void)ws_size;
  const float* x    = (const float*)d_in[0];
  const float* mu   = (const float*)d_in[1];
  const float* qw   = (const float*)d_in[2];
  const float* qb   = (const float*)d_in[3];
  const float* kw   = (const float*)d_in[4];
  const float* kb   = (const float*)d_in[5];
  const float* vw   = (const float*)d_in[6];
  const float* vb   = (const float*)d_in[7];
  const float* wih  = (const float*)d_in[8];
  const float* whh  = (const float*)d_in[9];
  const float* bih  = (const float*)d_in[10];
  const float* bhh  = (const float*)d_in[11];
  const float* niw  = (const float*)d_in[12];
  const float* nib  = (const float*)d_in[13];
  const float* nsw  = (const float*)d_in[14];
  const float* nsb  = (const float*)d_in[15];
  const float* nmw  = (const float*)d_in[16];
  const float* nmb  = (const float*)d_in[17];
  const float* fc1w = (const float*)d_in[18];
  const float* fc1b = (const float*)d_in[19];
  const float* fc2w = (const float*)d_in[20];
  const float* fc2b = (const float*)d_in[21];
  float* out = (float*)d_out;

  char* ws   = (char*)d_ws;
  size_t off = 0;
  auto alloc = [&](size_t bytes) -> void* {
    size_t a = (off + 255) & ~(size_t)255;
    off      = a + bytes;
    return (void*)(ws + a);
  };
  float*     mr      = (float*)alloc((size_t)XROWS * 2 * 4);
  _Float16*  qw16    = (_Float16*)alloc((size_t)CH * CH * 2);
  _Float16*  kw16    = (_Float16*)alloc((size_t)CH * CH * 2);
  _Float16*  vw16    = (_Float16*)alloc((size_t)CH * CH * 2);
  _Float16*  wih16   = (_Float16*)alloc((size_t)3 * CH * CH * 2);
  _Float16*  whh16   = (_Float16*)alloc((size_t)3 * CH * CH * 2);
  _Float16*  fc1w16  = (_Float16*)alloc((size_t)HID * CH * 2);
  _Float16*  fc2w16  = (_Float16*)alloc((size_t)CH * HID * 2);
  _Float16*  k16     = (_Float16*)alloc((size_t)XROWS * CH * 2);
  _Float16*  v16     = (_Float16*)alloc((size_t)XROWS * CH * 2);
  float*     slots   = (float*)alloc((size_t)ROWS * CH * 4);
  _Float16*  slots16 = (_Float16*)alloc((size_t)ROWS * CH * 2);
  _Float16*  sn16    = (_Float16*)alloc((size_t)ROWS * CH * 2);
  _Float16*  mn16    = (_Float16*)alloc((size_t)ROWS * CH * 2);
  _Float16*  qpad    = (_Float16*)alloc((size_t)BATCH * 16 * CH * 2);
  float*     attn    = (float*)alloc((size_t)BATCH * NSLOT * NTOK * 4);
  float*     rowsum  = (float*)alloc((size_t)ROWS * 4);
  float*     upd32   = (float*)alloc((size_t)ROWS * CH * 4);
  _Float16*  upd16   = (_Float16*)alloc((size_t)ROWS * CH * 2);
  float*     gi      = (float*)alloc((size_t)ROWS * 3 * CH * 4);
  float*     gh      = (float*)alloc((size_t)ROWS * 3 * CH * 4);
  _Float16*  h116    = (_Float16*)alloc((size_t)ROWS * HID * 2);

  auto cast = [&](const float* s, _Float16* d, int n) {
    k_cast_f32_f16<<<(n + 255) / 256, 256, 0, stream>>>(s, d, n);
  };
  cast(qw, qw16, CH * CH);
  cast(kw, kw16, CH * CH);
  cast(vw, vw16, CH * CH);
  cast(wih, wih16, 3 * CH * CH);
  cast(whh, whh16, 3 * CH * CH);
  cast(fc1w, fc1w16, HID * CH);
  cast(fc2w, fc2w16, CH * HID);

  k_zero_f16<<<(BATCH * 16 * CH + 255) / 256, 256, 0, stream>>>(qpad,
                                                               BATCH * 16 * CH);
  k_init_slots<<<(ROWS * CH + 255) / 256, 256, 0, stream>>>(mu, slots);
  k_row_stats<<<XROWS / 8, 256, 0, stream>>>(x, mr);
  k_proj_kv<<<dim3(XROWS / 128, CH / 128), 256, 0, stream>>>(
      x, mr, niw, nib, kw16, vw16, kb, vb, k16, v16);

  for (int it = 0; it < 3; ++it) {
    k_slot_ln<<<ROWS / 8, 256, 0, stream>>>(slots, nsw, nsb, sn16, slots16);
    k_gemm_xwT<<<dim3(ROWS / 128, CH / 128), 256, 0, stream>>>(
        sn16, qw16, qb, CH, CH, 4, (float*)nullptr, qpad);
    k_zero_f32<<<1, 256, 0, stream>>>(rowsum, ROWS);
    k_attn_logits<<<dim3(BATCH, NTOK / 128), 256, 0, stream>>>(qpad, k16, attn,
                                                               rowsum);
    k_attn_updates<<<dim3(BATCH, CH / 128), 256, 0, stream>>>(attn, rowsum, v16,
                                                              upd32, upd16);
    k_gemm_xwT<<<dim3(ROWS / 128, (3 * CH) / 128), 256, 0, stream>>>(
        upd16, wih16, bih, 3 * CH, CH, 2, gi, (_Float16*)nullptr);
    k_gemm_xwT<<<dim3(ROWS / 128, (3 * CH) / 128), 256, 0, stream>>>(
        slots16, whh16, bhh, 3 * CH, CH, 2, gh, (_Float16*)nullptr);
    k_gru<<<(ROWS * CH + 255) / 256, 256, 0, stream>>>(gi, gh, slots);
    k_slot_ln<<<ROWS / 8, 256, 0, stream>>>(slots, nmw, nmb, mn16, slots16);
    k_gemm_xwT<<<dim3(ROWS / 128, HID / 128), 256, 0, stream>>>(
        mn16, fc1w16, fc1b, HID, CH, 1, (float*)nullptr, h116);
    k_gemm_xwT<<<dim3(ROWS / 128, CH / 128), 256, 0, stream>>>(
        h116, fc2w16, fc2b, CH, HID, 3, slots, (_Float16*)nullptr);
  }
  k_maxpool<<<(BATCH * CH + 255) / 256, 256, 0, stream>>>(slots, out);
}
